// NbrAttn2_29532195127723
// MI455X (gfx1250) — matrix operand order repaired and checked by reading, not compiled, not run on hardware
//
#include <hip/hip_runtime.h>
#include <hip/hip_bf16.h>
#include <math.h>

typedef _Float16 half_t;
typedef __attribute__((ext_vector_type(16))) _Float16 v16h;
typedef __attribute__((ext_vector_type(8)))  float    v8f;

#define B_    4
#define T_    512
#define N_    16
#define D_    256
#define H_    8
#define DK_   32
#define CTS_  6
#define CN_   4
#define CE_   3
#define TSE_  192
#define AUXE_ 64
#define NT_   (N_ * T_)          // 8192
#define R_    (B_ * N_ * T_)     // 32768 rows for branch GEMMs
#define BT_   (B_ * T_)          // 2048

// ---------------------------------------------------------------------------
// WMMA fragment loaders (wave32, contiguous k order: k = 16*(lane/16) + element)
// ---------------------------------------------------------------------------

// A-matrix 16x32 f16: lanes 0-15 -> M=lane, halves {0..15}=K0..15
//                     lanes 16-31 -> M=lane-16, halves {0..15}=K16..31
__device__ __forceinline__ v16h load_a_frag(const half_t* __restrict__ A, int lda,
                                            int row0, int k0, int lane) {
  const int g = (lane >> 4) & 1;
  const int m = lane & 15;
  const half_t* p = A + (size_t)(row0 + m) * lda + k0 + 16 * g;
  v16h a;
#pragma unroll
  for (int h = 0; h < 8; ++h) a[h] = p[h];
#pragma unroll
  for (int h = 0; h < 8; ++h) a[8 + h] = p[8 + h];
  return a;
}

// B-matrix 32x16 f16 from TRANSPOSED weights Wt[N,K] row-major:
// lane (g,n) holds column n of B, halves h -> K = k0 + 16*g + h  (contiguous!)
__device__ __forceinline__ v16h load_bt_frag(const half_t* __restrict__ Wt, int K,
                                             int k0, int col0, int lane) {
  const int g = (lane >> 4) & 1;
  const int n = lane & 15;
  const half_t* p = Wt + (size_t)(col0 + n) * K + k0 + 16 * g;
  v16h b;
#pragma unroll
  for (int h = 0; h < 16; ++h) b[h] = p[h];
  return b;
}

__device__ __forceinline__ v8f wmma_f16(v16h a, v16h b, v8f c) {
  return __builtin_amdgcn_wmma_f32_16x16x32_f16(false, a, false, b, (short)0, c,
                                                false, false);
}

// ---------------------------------------------------------------------------
// Generic WMMA GEMM: out[M,N] = A[M,K](f16) * Wt[N,K]^T(f16) + bias
// Each wave computes a 32x64 macro-tile (2 M-tiles x 4 N-tiles, 8 WMMAs/kstep).
// Epilogue: f32 store (+residual, column offset) | f16 store | transposed f16
// store for V ([B, D, NT] layout). M%32==0, N%64==0, K%32==0.
// ---------------------------------------------------------------------------
__global__ void wmma_gemm_kernel(const half_t* __restrict__ A, int lda,
                                 const half_t* __restrict__ Wt,
                                 const float* __restrict__ bias,
                                 int M, int Nn, int K,
                                 float* __restrict__ outF, int ldoF, int coloff,
                                 const float* __restrict__ resid,
                                 half_t* __restrict__ outH, int ldoH,
                                 int transV) {
  const int lane = threadIdx.x & 31;
  const int wid  = threadIdx.x >> 5;
  const int tiles_n = Nn >> 6;
  const int total   = (M >> 5) * tiles_n;
  const int tile = blockIdx.x * (blockDim.x >> 5) + wid;
  if (tile >= total) return;
  const int mt = tile / tiles_n, nt = tile - mt * tiles_n;
  const int row0 = mt << 5, col0 = nt << 6;

  const int g = lane >> 4, n = lane & 15;

  v8f acc[2][4];
#pragma unroll
  for (int i = 0; i < 2; ++i)
#pragma unroll
    for (int j = 0; j < 4; ++j) acc[i][j] = (v8f){0.f,0.f,0.f,0.f,0.f,0.f,0.f,0.f};

  for (int k0 = 0; k0 < K; k0 += 32) {
    const v16h a0 = load_a_frag(A, lda, row0,      k0, lane);
    const v16h a1 = load_a_frag(A, lda, row0 + 16, k0, lane);
#pragma unroll
    for (int j = 0; j < 4; ++j) {
      const v16h b = load_bt_frag(Wt, K, k0, col0 + 16 * j, lane);
      acc[0][j] = wmma_f16(a0, b, acc[0][j]);
      acc[1][j] = wmma_f16(a1, b, acc[1][j]);
    }
  }

#pragma unroll
  for (int i = 0; i < 2; ++i) {
#pragma unroll
    for (int j = 0; j < 4; ++j) {
      const int cc = col0 + 16 * j + n;
      const float bv = bias ? bias[cc] : 0.f;
#pragma unroll
      for (int r = 0; r < 8; ++r) {
        const int row = row0 + 16 * i + r + 8 * g;
        const float v = acc[i][j][r] + bv;
        if (transV) {
          // transposed V store: vT[b][cc][key], NT_ = 8192 = 2^13
          const int bb  = row >> 13;
          const int key = row & (NT_ - 1);
          outH[((size_t)bb * D_ + cc) * NT_ + key] = (half_t)v;
        } else if (outH) {
          outH[(size_t)row * ldoH + cc] = (half_t)v;
        } else {
          float o = v;
          if (resid) o += resid[(size_t)row * ldoF + coloff + cc];
          outF[(size_t)row * ldoF + coloff + cc] = o;
        }
      }
    }
  }
}

// ---------------------------------------------------------------------------
// Sinusoidal positional-encoding table [T, D]
// ---------------------------------------------------------------------------
__global__ void pe_kernel(float* __restrict__ pe) {
  const int idx = blockIdx.x * blockDim.x + threadIdx.x;
  if (idx >= T_ * D_) return;
  const int t = idx / D_, d = idx - t * D_;
  const int i = d >> 1;
  const float div = expf((float)(2 * i) * (-logf(10000.f) / (float)D_));
  const float arg = (float)t * div;
  pe[idx] = (d & 1) ? cosf(arg) : sinf(arg);
}

// f32 [K,N] -> f16 transposed [N,K]
__global__ void cvt_t_kernel(const float* __restrict__ src, half_t* __restrict__ dst,
                             int K, int N) {
  const int i = blockIdx.x * blockDim.x + threadIdx.x;
  if (i >= K * N) return;
  const int k = i / N, n = i - k * N;
  dst[(size_t)n * K + k] = (half_t)src[i];
}

// ---------------------------------------------------------------------------
// xq = x + pe ; xn = LayerNorm(xq) * g + b  (one 256-thread block per row)
// ---------------------------------------------------------------------------
__global__ void xq_ln_kernel(const float* __restrict__ x, const float* __restrict__ pe,
                             const float* __restrict__ gam, const float* __restrict__ bet,
                             float* __restrict__ xq, half_t* __restrict__ xn) {
  __shared__ float red[256];
  const int row = blockIdx.x;       // b*T + t
  const int t = row & (T_ - 1);
  const int d = threadIdx.x;
  const float v = x[(size_t)row * D_ + d] + pe[t * D_ + d];
  xq[(size_t)row * D_ + d] = v;

  red[d] = v;
  __syncthreads();
  for (int s = 128; s > 0; s >>= 1) {
    if (d < s) red[d] += red[d + s];
    __syncthreads();
  }
  const float mu = red[0] * (1.f / D_);
  __syncthreads();
  const float c = v - mu;
  red[d] = c * c;
  __syncthreads();
  for (int s = 128; s > 0; s >>= 1) {
    if (d < s) red[d] += red[d + s];
    __syncthreads();
  }
  const float var = red[0] * (1.f / D_);
  const float inv = rsqrtf(var + 1e-6f);
  xn[(size_t)row * D_ + d] = (half_t)(c * inv * gam[d] + bet[d]);
}

// ---------------------------------------------------------------------------
// First-layer branch MLPs (tiny K: 6/4/3) with ReLU, output f16 hidden.
// One 256-thread block per (b,n,t) row. Input layout [B,N,C,T].
// ---------------------------------------------------------------------------
__global__ void branch1_kernel(const float* __restrict__ md, const float* __restrict__ na,
                               const float* __restrict__ ea,
                               const float* __restrict__ Wts1, const float* __restrict__ bts1,
                               const float* __restrict__ Wa1,  const float* __restrict__ ba1,
                               const float* __restrict__ We1,  const float* __restrict__ be1,
                               half_t* __restrict__ h_ts, half_t* __restrict__ h_aux,
                               half_t* __restrict__ h_edge) {
  const int row = blockIdx.x;          // (b*N + n)*T + t
  const int t  = row & (T_ - 1);
  const int bn = row >> 9;             // T_ = 512
  const int j = threadIdx.x;
  const float* mdp = md + (size_t)bn * CTS_ * T_ + t;
  const float* nap = na + (size_t)bn * CN_ * T_ + t;
  const float* eap = ea + (size_t)bn * CE_ * T_ + t;

  if (j < TSE_) {
    float acc = bts1[j];
#pragma unroll
    for (int c = 0; c < CTS_; ++c) acc += mdp[c * T_] * Wts1[c * TSE_ + j];
    h_ts[(size_t)row * TSE_ + j] = (half_t)fmaxf(acc, 0.f);
  } else {
    const int jj = j - TSE_;
    float acc = ba1[jj];
#pragma unroll
    for (int c = 0; c < CN_; ++c) acc += nap[c * T_] * Wa1[c * AUXE_ + jj];
    h_aux[(size_t)row * AUXE_ + jj] = (half_t)fmaxf(acc, 0.f);
  }
  {
    float acc = be1[j];
#pragma unroll
    for (int c = 0; c < CE_; ++c) acc += eap[c * T_] * We1[c * D_ + j];
    h_edge[(size_t)row * D_ + j] = (half_t)fmaxf(acc, 0.f);
  }
}

// ---------------------------------------------------------------------------
// keys = f16(nbr*edge + pe), vals = f16(nbr + pe)
// ---------------------------------------------------------------------------
__global__ void combine_kv_kernel(const float* __restrict__ nbr, const float* __restrict__ edge,
                                  const float* __restrict__ pe,
                                  half_t* __restrict__ keys, half_t* __restrict__ vals) {
  const size_t idx = (size_t)blockIdx.x * blockDim.x + threadIdx.x;
  if (idx >= (size_t)R_ * D_) return;
  const int d = (int)(idx & (D_ - 1));
  const int t = (int)((idx >> 8) & (T_ - 1));   // D_ = 256
  const float p = pe[t * D_ + d];
  const float nv = nbr[idx];
  const float ev = edge[idx];
  keys[idx] = (half_t)(nv * ev + p);
  vals[idx] = (half_t)(nv + p);
}

// ---------------------------------------------------------------------------
// Flash cross-attention. One wave per (b, h, 16-query tile); streams 32 keys
// per iteration: 2 score WMMAs + online softmax + 2 P*V WMMAs.
// K is [B,NT,D] (row per key -> contiguous 32B/lane); V is TRANSPOSED
// vT[B,D,NT] so P*V B-fragments are also contiguous 32B/lane.
// ---------------------------------------------------------------------------
__global__ __launch_bounds__(32) void attn_kernel(
    const half_t* __restrict__ qh, const half_t* __restrict__ kh,
    const half_t* __restrict__ vT, const int* __restrict__ mask,
    half_t* __restrict__ ctx) {
  __shared__ float  sS[16][32];
  __shared__ half_t sP[16][32];
  __shared__ float  sAl[16];
  __shared__ float  sInv[16];

  const int lane = threadIdx.x & 31;
  const int blk = blockIdx.x;          // b*(H*T/16) + h*(T/16) + qt
  const int qt  = blk & 31;            // T/16 == 32
  const int hh_ = (blk >> 5) & 7;
  const int b   = blk >> 8;

  const int g  = lane >> 4;
  const int ln = lane & 15;

  // q A-fragment lives in registers for the whole KV sweep
  const v16h qa = load_a_frag(qh, D_, b * T_ + qt * 16, hh_ * DK_, lane);

  v8f acc0 = {0.f, 0.f, 0.f, 0.f, 0.f, 0.f, 0.f, 0.f};
  v8f acc1 = {0.f, 0.f, 0.f, 0.f, 0.f, 0.f, 0.f, 0.f};
  float m_i = -3.0e38f, l_i = 0.f;
  const float scale = 0.17677669529663688f;   // 1/sqrt(DK)

  const size_t kbase  = (size_t)b * NT_ * D_ + hh_ * DK_;
  const half_t* vTb   = vT + (size_t)(b * D_ + hh_ * DK_) * NT_;

  for (int kv0 = 0; kv0 < NT_; kv0 += 32) {
    // ---- scores for keys kv0..kv0+31 : two WMMAs (B = K^T tile) ----
#pragma unroll
    for (int jt = 0; jt < 2; ++jt) {
      const half_t* kp = kh + kbase + (size_t)(kv0 + 16 * jt + ln) * D_ + 16 * g;
      v16h bf;
#pragma unroll
      for (int u = 0; u < 16; ++u) bf[u] = kp[u];      // contiguous 32B per lane
      v8f z = {0.f, 0.f, 0.f, 0.f, 0.f, 0.f, 0.f, 0.f};
      v8f s = wmma_f16(qa, bf, z);
#pragma unroll
      for (int r = 0; r < 8; ++r) sS[r + 8 * g][ln + 16 * jt] = s[r];
    }
    __syncthreads();

    // ---- online softmax: lanes 0-15 each own one query row ----
    if (lane < 16) {
      const int4* mp = (const int4*)(mask + (size_t)(b * T_ + qt * 16 + lane) * NT_ + kv0);
      float sv[32];
      float mmax = m_i;
#pragma unroll
      for (int c4 = 0; c4 < 8; ++c4) {
        const int4 mm = mp[c4];
        const int mi[4] = {mm.x, mm.y, mm.z, mm.w};
#pragma unroll
        for (int u = 0; u < 4; ++u) {
          const int c = c4 * 4 + u;
          float s = sS[lane][c] * scale;
          if (mi[u] == 0) s = -1e9f;
          sv[c] = s;
          mmax = fmaxf(mmax, s);
        }
      }
      const float alpha = __expf(m_i - mmax);
      float psum = 0.f;
#pragma unroll
      for (int c = 0; c < 32; ++c) {
        const float p = __expf(sv[c] - mmax);
        psum += p;
        sP[lane][c] = (half_t)p;
      }
      l_i = l_i * alpha + psum;
      m_i = mmax;
      sAl[lane] = alpha;
    }
    __syncthreads();

    // ---- rescale accumulators by alpha(row) ----
#pragma unroll
    for (int r = 0; r < 8; ++r) {
      const float al = sAl[r + 8 * g];
      acc0[r] *= al;
      acc1[r] *= al;
    }

    // ---- P as A-fragment (two contiguous 16B LDS runs per lane) ----
    v16h pf;
#pragma unroll
    for (int u = 0; u < 8; ++u)  pf[u]     = sP[ln][u + 16 * g];
#pragma unroll
    for (int u = 0; u < 8; ++u)  pf[8 + u] = sP[ln][8 + u + 16 * g];

    // ---- V B-fragments from vT: contiguous 32B per lane ----
#pragma unroll
    for (int jt = 0; jt < 2; ++jt) {
      const half_t* vp = vTb + (size_t)(16 * jt + ln) * NT_ + kv0 + 16 * g;
      v16h vf;
#pragma unroll
      for (int u = 0; u < 16; ++u) vf[u] = vp[u];
      if (jt == 0) acc0 = wmma_f16(pf, vf, acc0);
      else         acc1 = wmma_f16(pf, vf, acc1);
    }
    __syncthreads();
  }

  if (lane < 16) sInv[lane] = 1.f / l_i;
  __syncthreads();

  const size_t obase = (size_t)(b * T_ + qt * 16) * D_ + hh_ * DK_;
#pragma unroll
  for (int r = 0; r < 8; ++r) {
    const float inv = sInv[r + 8 * g];
    ctx[obase + (size_t)(r + 8 * g) * D_ + ln]      = (half_t)(acc0[r] * inv);
    ctx[obase + (size_t)(r + 8 * g) * D_ + 16 + ln] = (half_t)(acc1[r] * inv);
  }
}

// ---------------------------------------------------------------------------
// host side
// ---------------------------------------------------------------------------
static inline void launch_gemm(hipStream_t stream,
                               const half_t* A, int lda, const half_t* Wt,
                               const float* bias, int M, int N, int K,
                               float* outF, int ldoF, int coloff, const float* resid,
                               half_t* outH, int ldoH, int transV) {
  const int waves = (M / 32) * (N / 64);
  const int blocks = (waves + 7) / 8;
  wmma_gemm_kernel<<<blocks, 256, 0, stream>>>(A, lda, Wt, bias, M, N, K,
                                               outF, ldoF, coloff, resid, outH, ldoH,
                                               transV);
}

extern "C" void kernel_launch(void* const* d_in, const int* in_sizes, int n_in,
                              void* d_out, int out_size, void* d_ws, size_t ws_size,
                              hipStream_t stream) {
  const float* x    = (const float*)d_in[0];
  const float* md   = (const float*)d_in[1];
  const float* na   = (const float*)d_in[2];
  const float* ea   = (const float*)d_in[3];
  const int*   msk  = (const int*)d_in[4];
  const float* Wts1 = (const float*)d_in[5];  const float* bts1 = (const float*)d_in[6];
  const float* Wts2 = (const float*)d_in[7];  const float* bts2 = (const float*)d_in[8];
  const float* Wa1  = (const float*)d_in[9];  const float* ba1  = (const float*)d_in[10];
  const float* Wa2  = (const float*)d_in[11]; const float* ba2  = (const float*)d_in[12];
  const float* We1  = (const float*)d_in[13]; const float* be1  = (const float*)d_in[14];
  const float* We2  = (const float*)d_in[15]; const float* be2  = (const float*)d_in[16];
  const float* Wq   = (const float*)d_in[17]; const float* bq   = (const float*)d_in[18];
  const float* Wk   = (const float*)d_in[19]; const float* bk   = (const float*)d_in[20];
  const float* Wv   = (const float*)d_in[21]; const float* bv   = (const float*)d_in[22];
  const float* Wo   = (const float*)d_in[23]; const float* bo   = (const float*)d_in[24];
  const float* lng  = (const float*)d_in[25]; const float* lnb  = (const float*)d_in[26];
  float* out = (float*)d_out;

  char* w = (char*)d_ws;
  size_t off = 0;
  auto alloc = [&](size_t bytes) -> void* {
    void* p = w + off;
    off += (bytes + 255) & ~(size_t)255;
    return p;
  };

  float*  pe     = (float*)alloc(sizeof(float) * T_ * D_);
  float*  xq     = (float*)alloc(sizeof(float) * (size_t)BT_ * D_);
  half_t* xn     = (half_t*)alloc(sizeof(half_t) * (size_t)BT_ * D_);
  half_t* h_ts   = (half_t*)alloc(sizeof(half_t) * (size_t)R_ * TSE_);
  half_t* h_aux  = (half_t*)alloc(sizeof(half_t) * (size_t)R_ * AUXE_);
  half_t* h_edge = (half_t*)alloc(sizeof(half_t) * (size_t)R_ * D_);
  float*  nbr    = (float*)alloc(sizeof(float) * (size_t)R_ * D_);
  float*  edge   = (float*)alloc(sizeof(float) * (size_t)R_ * D_);
  half_t* keysh  = (half_t*)alloc(sizeof(half_t) * (size_t)R_ * D_);
  half_t* valsh  = (half_t*)alloc(sizeof(half_t) * (size_t)R_ * D_);
  half_t* qh     = (half_t*)alloc(sizeof(half_t) * (size_t)BT_ * D_);
  half_t* kh     = (half_t*)alloc(sizeof(half_t) * (size_t)R_ * D_);
  half_t* vTh    = (half_t*)alloc(sizeof(half_t) * (size_t)B_ * D_ * NT_);
  half_t* ctxh   = (half_t*)alloc(sizeof(half_t) * (size_t)BT_ * D_);
  half_t* Wts2t  = (half_t*)alloc(sizeof(half_t) * TSE_ * TSE_);
  half_t* Wa2t   = (half_t*)alloc(sizeof(half_t) * AUXE_ * AUXE_);
  half_t* We2t   = (half_t*)alloc(sizeof(half_t) * D_ * D_);
  half_t* Wqt    = (half_t*)alloc(sizeof(half_t) * D_ * D_);
  half_t* Wkt    = (half_t*)alloc(sizeof(half_t) * D_ * D_);
  half_t* Wvt    = (half_t*)alloc(sizeof(half_t) * D_ * D_);
  half_t* Wot    = (half_t*)alloc(sizeof(half_t) * D_ * D_);
  (void)ws_size; (void)in_sizes; (void)n_in; (void)out_size;

  // PE table
  pe_kernel<<<(T_ * D_ + 255) / 256, 256, 0, stream>>>(pe);

  // f32 -> f16 transposed weight conversion ([K,N] -> [N,K])
  cvt_t_kernel<<<(TSE_ * TSE_ + 255) / 256, 256, 0, stream>>>(Wts2, Wts2t, TSE_, TSE_);
  cvt_t_kernel<<<(AUXE_ * AUXE_ + 255) / 256, 256, 0, stream>>>(Wa2, Wa2t, AUXE_, AUXE_);
  cvt_t_kernel<<<(D_ * D_ + 255) / 256, 256, 0, stream>>>(We2, We2t, D_, D_);
  cvt_t_kernel<<<(D_ * D_ + 255) / 256, 256, 0, stream>>>(Wq, Wqt, D_, D_);
  cvt_t_kernel<<<(D_ * D_ + 255) / 256, 256, 0, stream>>>(Wk, Wkt, D_, D_);
  cvt_t_kernel<<<(D_ * D_ + 255) / 256, 256, 0, stream>>>(Wv, Wvt, D_, D_);
  cvt_t_kernel<<<(D_ * D_ + 255) / 256, 256, 0, stream>>>(Wo, Wot, D_, D_);

  // xq = x + pe ; xn = LN(xq)
  xq_ln_kernel<<<BT_, 256, 0, stream>>>(x, pe, lng, lnb, xq, xn);

  // first-layer branch MLPs (tiny K) -> f16 hidden activations
  branch1_kernel<<<R_, 256, 0, stream>>>(md, na, ea, Wts1, bts1, Wa1, ba1, We1, be1,
                                         h_ts, h_aux, h_edge);

  // second-layer GEMMs (WMMA): nbr = [ts2 | a2], edge = e2
  launch_gemm(stream, h_ts, TSE_, Wts2t, bts2, R_, TSE_, TSE_,
              nbr, D_, 0, nullptr, nullptr, 0, 0);
  launch_gemm(stream, h_aux, AUXE_, Wa2t, ba2, R_, AUXE_, AUXE_,
              nbr, D_, TSE_, nullptr, nullptr, 0, 0);
  launch_gemm(stream, h_edge, D_, We2t, be2, R_, D_, D_,
              edge, D_, 0, nullptr, nullptr, 0, 0);

  // keys/vals combine with PE
  combine_kv_kernel<<<(int)(((size_t)R_ * D_ + 255) / 256), 256, 0, stream>>>(
      nbr, edge, pe, keysh, valsh);

  // Q/K/V projections (WMMA, f16 outputs; V stored transposed [B,D,NT])
  launch_gemm(stream, xn, D_, Wqt, bq, BT_, D_, D_, nullptr, 0, 0, nullptr, qh, D_, 0);
  launch_gemm(stream, keysh, D_, Wkt, bk, R_, D_, D_, nullptr, 0, 0, nullptr, kh, D_, 0);
  launch_gemm(stream, valsh, D_, Wvt, bv, R_, D_, D_, nullptr, 0, 0, nullptr, vTh, 0, 1);

  // flash cross-attention (WMMA scores + WMMA P*V)
  attn_kernel<<<B_ * H_ * (T_ / 16), 32, 0, stream>>>(qh, kh, vTh, msk, ctxh);

  // output projection + bias + residual(xq) -> d_out (f32)
  launch_gemm(stream, ctxh, D_, Wot, bo, BT_, D_, D_, out, D_, 0, xq, nullptr, 0, 0);
}
